// Attention_71390946394453
// MI455X (gfx1250) — matrix operand order repaired and checked by reading, not compiled, not run on hardware
//
#include <hip/hip_runtime.h>
#include <hip/hip_bf16.h>

// ---------------------------------------------------------------------------
// CDNA5 (gfx1250) wave32 WMMA attention:
//   qkv = x @ Wqkv^T            (bf16 WMMA, f32 accum)
//   rotary(q,k) + split to [B,H,T,hd] bf16
//   flash attention (bf16 WMMA, f32 accum, online softmax, async LDS staging)
//   out = O @ Wout^T            (bf16 WMMA, f32 accum)
// ---------------------------------------------------------------------------

typedef __attribute__((ext_vector_type(16))) __bf16 v16bf;
typedef __attribute__((ext_vector_type(8)))  float  v8f;

union FragAB { v16bf v; unsigned int u[8]; __bf16 h[16]; };

__device__ inline v8f zero8() {
  v8f z = {0.f, 0.f, 0.f, 0.f, 0.f, 0.f, 0.f, 0.f};
  return z;
}

#if defined(__has_builtin)
#if __has_builtin(__builtin_amdgcn_global_load_async_to_lds_b128)
#define HAVE_ASYNC_LDS 1
#endif
#endif

#ifdef HAVE_ASYNC_LDS
// Exact parameter types per hipcc diagnostic:
//   arg0: int __attribute__((vector_size(16))) __device__ * (AS1 / global)
//   arg1: same pointee in __shared__ (AS3 / LDS)
typedef int v4i_raw __attribute__((vector_size(16)));
typedef __attribute__((address_space(1))) v4i_raw* gptr_v4i;
typedef __attribute__((address_space(3))) v4i_raw* lptr_v4i;
#endif

__device__ inline void wait_asynccnt0() {
#if defined(__has_builtin)
#if __has_builtin(__builtin_amdgcn_s_wait_asynccnt)
  __builtin_amdgcn_s_wait_asynccnt(0);
  return;
#endif
#endif
#ifdef HAVE_ASYNC_LDS
  asm volatile("s_wait_asynccnt 0x0" ::: "memory");
#endif
}

// ---------------------------------------------------------------------------
// GEMM (NT): C[M,N] = A[M,K] * B[N,K]^T, fp32 in/out, bf16 WMMA accum f32.
// Block: 256 threads = 8 waves, tile 128(M) x 64(N), K step 32.
// Wave grid 4(M) x 2(N); each wave owns 2x2 WMMA 16x16 tiles.
// Staging: float4 loads into independent regs -> packed bf16x2 -> ds_store_b64.
// ---------------------------------------------------------------------------
__global__ __launch_bounds__(256)
void gemm_nt_kernel(const float* __restrict__ A, const float* __restrict__ B,
                    float* __restrict__ C, int M, int N, int K) {
  __shared__ __bf16 As[128][36];   // 32 + 4 pad (keeps b64 stores 8B aligned)
  __shared__ __bf16 Bs[64][36];

  const int tid  = threadIdx.x;
  const int lane = tid & 31;
  const int wave = tid >> 5;
  const int half = lane >> 4;
  const int ln16 = lane & 15;

  const int mBlock = blockIdx.y * 128;
  const int nBlock = blockIdx.x * 64;
  const int waveM  = wave >> 1;   // 0..3
  const int waveN  = wave & 1;    // 0..1

  v8f acc[2][2];
#pragma unroll
  for (int i = 0; i < 2; ++i)
#pragma unroll
    for (int j = 0; j < 2; ++j) acc[i][j] = zero8();

  for (int k0 = 0; k0 < K; k0 += 32) {
    __syncthreads();
    // --- Stage A (128x32) and B (64x32): wide loads first (MLP), then cvt+store.
    float4 ra[4];
#pragma unroll
    for (int i = 0; i < 4; ++i) {
      int ci = tid + i * 256;              // 1024 float4 chunks
      int r = ci >> 3, c4 = (ci & 7) * 4;
      ra[i] = *(const float4*)&A[(size_t)(mBlock + r) * K + k0 + c4];
    }
    float4 rb[2];
#pragma unroll
    for (int i = 0; i < 2; ++i) {
      int ci = tid + i * 256;              // 512 float4 chunks
      int r = ci >> 3, c4 = (ci & 7) * 4;
      rb[i] = *(const float4*)&B[(size_t)(nBlock + r) * K + k0 + c4];
    }
#pragma unroll
    for (int i = 0; i < 4; ++i) {
      int ci = tid + i * 256;
      int r = ci >> 3, c4 = (ci & 7) * 4;
      union { uint2 u2; __bf16 h[4]; } pk;
      pk.h[0] = (__bf16)ra[i].x; pk.h[1] = (__bf16)ra[i].y;
      pk.h[2] = (__bf16)ra[i].z; pk.h[3] = (__bf16)ra[i].w;
      *(uint2*)&As[r][c4] = pk.u2;
    }
#pragma unroll
    for (int i = 0; i < 2; ++i) {
      int ci = tid + i * 256;
      int r = ci >> 3, c4 = (ci & 7) * 4;
      union { uint2 u2; __bf16 h[4]; } pk;
      pk.h[0] = (__bf16)rb[i].x; pk.h[1] = (__bf16)rb[i].y;
      pk.h[2] = (__bf16)rb[i].z; pk.h[3] = (__bf16)rb[i].w;
      *(uint2*)&Bs[r][c4] = pk.u2;
    }
    // Prefetch next K tile (global_prefetch_b8).
    if (k0 + 32 < K) {
      __builtin_prefetch(&A[(size_t)(mBlock + (tid >> 1)) * K + k0 + 32 + (tid & 1) * 16], 0, 1);
      __builtin_prefetch(&B[(size_t)(nBlock + (tid & 63)) * K + k0 + 32 + ((tid >> 6) & 1) * 16], 0, 1);
    }
    __syncthreads();

    // --- Fragments (packed 2xbf16 dword reads from LDS) and WMMA.
    FragAB a[2], b[2];
#pragma unroll
    for (int i = 0; i < 2; ++i) {
      int mRow = waveM * 32 + i * 16 + ln16;
#pragma unroll
      for (int j = 0; j < 8; ++j) {
        int kk = (j & 3) * 2 + half * 8 + (j >> 2) * 16;   // A-layout K index
        a[i].u[j] = *(const unsigned int*)&As[mRow][kk];
      }
    }
#pragma unroll
    for (int i = 0; i < 2; ++i) {
      int nRow = waveN * 32 + i * 16 + ln16;
#pragma unroll
      for (int j = 0; j < 8; ++j) {
        int kk = (j & 3) * 2 + half * 8 + (j >> 2) * 16;   // B-layout K index
        b[i].u[j] = *(const unsigned int*)&Bs[nRow][kk];
      }
    }
#pragma unroll
    for (int i = 0; i < 2; ++i)
#pragma unroll
      for (int jn = 0; jn < 2; ++jn)
        acc[i][jn] = __builtin_amdgcn_wmma_f32_16x16x32_bf16(
            false, a[i].v, false, b[jn].v, (short)0, acc[i][jn], false, false);
  }

  // Epilogue: C/D layout lane=N col, VGPR r -> row r + 8*half.
#pragma unroll
  for (int i = 0; i < 2; ++i) {
#pragma unroll
    for (int jn = 0; jn < 2; ++jn) {
      int nCol = nBlock + waveN * 32 + jn * 16 + ln16;
#pragma unroll
      for (int r = 0; r < 8; ++r) {
        int mRow = mBlock + waveM * 32 + i * 16 + r + half * 8;
        C[(size_t)mRow * N + nCol] = acc[i][jn][r];
      }
    }
  }
}

// ---------------------------------------------------------------------------
// Rotary + split: qkv[B,T,3,H,hd] fp32 -> Q,K (rotary), V as bf16 [B,H,T,hd].
// One thread per (b,t,h,d2), d2 in [0,32).
// ---------------------------------------------------------------------------
__global__ __launch_bounds__(256)
void rotary_split_kernel(const float* __restrict__ qkv,
                         const float* __restrict__ freqs,
                         __bf16* __restrict__ Qp, __bf16* __restrict__ Kp,
                         __bf16* __restrict__ Vp) {
  const int T = 2048, H = 16;
  int idx = blockIdx.x * 256 + threadIdx.x;   // B*T*H*32 = 2^21 threads
  int d2 = idx & 31;
  int h  = (idx >> 5) & 15;
  int t  = (idx >> 9) & 2047;
  int b  = idx >> 20;

  size_t row = (size_t)(b * T + t) * 3072;
  float2 q12 = *(const float2*)&qkv[row + h * 64 + 2 * d2];
  float2 k12 = *(const float2*)&qkv[row + 1024 + h * 64 + 2 * d2];
  float2 v12 = *(const float2*)&qkv[row + 2048 + h * 64 + 2 * d2];

  float fr = freqs[t * 32 + d2];
  float sn = __sinf(fr), cs = __cosf(fr);

  size_t obase = ((size_t)(b * H + h) * T + t) * 64;

  Qp[obase + d2]      = (__bf16)(q12.x * cs - q12.y * sn);
  Qp[obase + 32 + d2] = (__bf16)(q12.x * sn + q12.y * cs);
  Kp[obase + d2]      = (__bf16)(k12.x * cs - k12.y * sn);
  Kp[obase + 32 + d2] = (__bf16)(k12.x * sn + k12.y * cs);

  union { unsigned int u; __bf16 h2[2]; } pv;
  pv.h2[0] = (__bf16)v12.x; pv.h2[1] = (__bf16)v12.y;
  *(unsigned int*)&Vp[obase + 2 * d2] = pv.u;
}

// ---------------------------------------------------------------------------
// Flash attention: block = 4 waves = 128 threads, one (b,h) x 64-row q tile.
// K/V tiles (32 keys x 64) staged row-major in LDS via async global->LDS DMA
// (ASYNCcnt) when available; vectorized synchronous copy otherwise.
// ---------------------------------------------------------------------------
__global__ __launch_bounds__(128)
void attn_kernel(const __bf16* __restrict__ Qp, const __bf16* __restrict__ Kp,
                 const __bf16* __restrict__ Vp, float* __restrict__ O) {
  const int T = 2048, HD = 64, H = 16;
  __shared__ __bf16 Ks[32][72];       // stride 72: 16B-aligned chunks, bank-spread
  __shared__ __bf16 Vs[32][72];
  __shared__ __bf16 Ps[4][16][34];    // per-wave P scratch 16x32

  const int tid  = threadIdx.x;
  const int lane = tid & 31;
  const int wave = tid >> 5;
  const int half = lane >> 4;
  const int ln16 = lane & 15;

  const int bh    = blockIdx.y;            // b*H + h
  const int qtile = blockIdx.x;            // 0..T/64-1
  const int qBase = qtile * 64 + wave * 16;

  const size_t base = (size_t)bh * T * HD;

  // Persistent Q fragments (two A-frags covering hd = 64).
  FragAB aQ[2];
#pragma unroll
  for (int kk = 0; kk < 2; ++kk) {
    int m = qBase + ln16;
#pragma unroll
    for (int j = 0; j < 8; ++j) {
      int d = (j & 3) * 2 + half * 8 + (j >> 2) * 16 + kk * 32;
      aQ[kk].u[j] = *(const unsigned int*)(Qp + base + (size_t)m * HD + d);
    }
  }

  v8f o[4];
#pragma unroll
  for (int i = 0; i < 4; ++i) o[i] = zero8();
  float mrun[8], lrun[8];
#pragma unroll
  for (int r = 0; r < 8; ++r) { mrun[r] = -1.0e30f; lrun[r] = 0.0f; }

  const int nTiles = qtile * 2 + 2;        // causal bound for this block
  for (int kt = 0; kt < nTiles; ++kt) {
    const int kBase = kt * 32;
    __syncthreads();                       // WAR guard on Ks/Vs
    // --- Stage K and V tiles: 256 x 16B chunks each, split across 128 threads.
#pragma unroll
    for (int i = 0; i < 2; ++i) {
      int ci = tid + i * 128;              // chunk 0..255
      int r  = ci >> 3;                    // key row 0..31
      int c  = (ci & 7) * 8;               // d col (multiple of 8)
      const __bf16* gk = Kp + base + (size_t)(kBase + r) * HD + c;
      const __bf16* gv = Vp + base + (size_t)(kBase + r) * HD + c;
#ifdef HAVE_ASYNC_LDS
      __builtin_amdgcn_global_load_async_to_lds_b128(
          (gptr_v4i)(void*)gk, (lptr_v4i)(void*)&Ks[r][c], 0, 0);
      __builtin_amdgcn_global_load_async_to_lds_b128(
          (gptr_v4i)(void*)gv, (lptr_v4i)(void*)&Vs[r][c], 0, 0);
#else
      uint4 k4 = *(const uint4*)gk;
      uint4 v4 = *(const uint4*)gv;
      *(uint4*)&Ks[r][c] = k4;
      *(uint4*)&Vs[r][c] = v4;
#endif
    }
#ifdef HAVE_ASYNC_LDS
    wait_asynccnt0();
#endif
    __syncthreads();

    // --- S = Q * K^T  (two 16x16 score tiles over 32 keys)
    v8f s[2] = {zero8(), zero8()};
#pragma unroll
    for (int sub = 0; sub < 2; ++sub) {
      int nk = sub * 16 + ln16;
#pragma unroll
      for (int kk = 0; kk < 2; ++kk) {
        FragAB bk;
#pragma unroll
        for (int j = 0; j < 8; ++j)
          bk.u[j] = *(const unsigned int*)&Ks[nk][(j & 3) * 2 + half * 8 + (j >> 2) * 16 + kk * 32];
        s[sub] = __builtin_amdgcn_wmma_f32_16x16x32_bf16(
            false, aQ[kk].v, false, bk.v, (short)0, s[sub], false, false);
      }
    }

    // --- Online softmax (row m = r + 8*half, cols across 16 lanes).
    const float sc = 0.125f;               // 1/sqrt(64)
#pragma unroll
    for (int r = 0; r < 8; ++r) {
      int qi = qBase + r + half * 8;
      float v0 = (kBase + ln16      <= qi) ? s[0][r] * sc : -1.0e30f;
      float v1 = (kBase + 16 + ln16 <= qi) ? s[1][r] * sc : -1.0e30f;
      float mx = fmaxf(v0, v1);
#pragma unroll
      for (int off = 8; off >= 1; off >>= 1) mx = fmaxf(mx, __shfl_xor(mx, off, 16));
      float nm = fmaxf(mrun[r], mx);
      float scale = __expf(mrun[r] - nm);
      float p0 = __expf(v0 - nm);
      float p1 = __expf(v1 - nm);
      float rs = p0 + p1;
#pragma unroll
      for (int off = 8; off >= 1; off >>= 1) rs += __shfl_xor(rs, off, 16);
      lrun[r] = lrun[r] * scale + rs;
      mrun[r] = nm;
#pragma unroll
      for (int dsub = 0; dsub < 4; ++dsub) o[dsub][r] *= scale;
      // Spill P (C-layout) to LDS so it can be re-read in A-layout.
      Ps[wave][r + half * 8][ln16]      = (__bf16)p0;
      Ps[wave][r + half * 8][16 + ln16] = (__bf16)p1;
    }

    // Same-wave LDS RAW: DS ops are in-order per wave; wait to be explicit.
    asm volatile("s_wait_dscnt 0x0" ::: "memory");

    FragAB aP;
#pragma unroll
    for (int j = 0; j < 8; ++j) {
      int kcol = j * 2 + half * 16;
      aP.u[j] = *(const unsigned int*)&Ps[wave][ln16][kcol];
    }

    // --- O += P * V (4 hd subtiles). B-frag gathered from row-major Vs.
#pragma unroll
    for (int dsub = 0; dsub < 4; ++dsub) {
      FragAB bv;
      int d = dsub * 16 + ln16;
#pragma unroll
      for (int e = 0; e < 16; ++e)
        bv.h[e] = Vs[e + half * 16][d];    // k = e + half*16
      o[dsub] = __builtin_amdgcn_wmma_f32_16x16x32_bf16(
          false, aP.v, false, bv.v, (short)0, o[dsub], false, false);
    }
  }

  // --- Epilogue: normalize, scatter to [B, T, H*hd] fp32 (bqhd layout).
  const int b = bh / H, h = bh % H;
#pragma unroll
  for (int r = 0; r < 8; ++r) {
    float inv = 1.0f / lrun[r];
    int tq = qBase + r + half * 8;
#pragma unroll
    for (int dsub = 0; dsub < 4; ++dsub) {
      int d = dsub * 16 + ln16;
      O[(((size_t)b * T + tq) * H + h) * HD + d] = o[dsub][r] * inv;
    }
  }
}

// ---------------------------------------------------------------------------
// Host-side launch
// ---------------------------------------------------------------------------
extern "C" void kernel_launch(void* const* d_in, const int* in_sizes, int n_in,
                              void* d_out, int out_size, void* d_ws, size_t ws_size,
                              hipStream_t stream) {
  const float* x     = (const float*)d_in[0];   // [2,2048,1024]
  const float* freqs = (const float*)d_in[1];   // [2048,32]
  const float* Wqkv  = (const float*)d_in[2];   // [3072,1024]
  const float* Wout  = (const float*)d_in[3];   // [1024,1024]
  float* out = (float*)d_out;                   // [2,2048,1024]

  const int M = 4096;      // B*T
  const int C = 1024;
  const int N1 = 3072;

  char* ws = (char*)d_ws;
  float*  qkv = (float*)ws;                               // 4096*3072 f32 = 48 MB
  __bf16* Qp  = (__bf16*)(ws + (size_t)M * N1 * 4);       // 4M bf16 = 8 MB each
  __bf16* Kp  = Qp + (size_t)4194304;
  __bf16* Vp  = Kp + (size_t)4194304;
  float*  O   = qkv;   // qkv region is dead after rotary_split; reuse for O

  // 1) qkv = x @ Wqkv^T
  gemm_nt_kernel<<<dim3(N1 / 64, M / 128), 256, 0, stream>>>(x, Wqkv, qkv, M, N1, C);

  // 2) rotary + split to bf16 [B,H,T,hd]
  rotary_split_kernel<<<8192, 256, 0, stream>>>(qkv, freqs, Qp, Kp, Vp);

  // 3) flash attention -> O [B,T,C] fp32
  attn_kernel<<<dim3(32, 32), 128, 0, stream>>>(Qp, Kp, Vp, O);

  // 4) out = O @ Wout^T
  gemm_nt_kernel<<<dim3(C / 64, M / 128), 256, 0, stream>>>(O, Wout, out, M, C, C);
}
